// LigerFusedLinearKTOLoss_50972671869340
// MI455X (gfx1250) — compile-verified, matrix operand order checked by reading, not run on hardware
//
#include <hip/hip_runtime.h>
#include <hip/hip_bf16.h>

#define IGNORE_INDEX (-100)
#define KTO_BETA 0.1f

typedef __bf16 bf16_t;
typedef __attribute__((ext_vector_type(16))) __bf16 v16bf;
typedef __attribute__((ext_vector_type(8)))  __bf16 v8bf;
typedef __attribute__((ext_vector_type(8)))  float  v8f;
typedef __attribute__((ext_vector_type(4)))  float  f4;

// Problem constants (from reference setup_inputs)
static constexpr int BATCH  = 4;
static constexpr int SEQ    = 512;
static constexpr int HID    = 1024;        // K dimension
static constexpr int VOCAB  = 32000;       // N dimension
static constexpr int MROWS  = BATCH * SEQ; // 2048

// Tiling
static constexpr int M_TILE     = 32;                 // rows per workgroup
static constexpr int WAVES      = 16;                 // 512 threads, wave32
static constexpr int NPW        = 16;                 // columns per wave per iter
static constexpr int N_PER_ITER = WAVES * NPW;        // 256
static constexpr int N_ITERS    = VOCAB / N_PER_ITER; // 125 (exact)
static constexpr int VSPLIT     = 5;                  // vocab split across WGs
static constexpr int ITERS_PER_SPLIT = N_ITERS / VSPLIT; // 25 (exact)

// LDS row stride for A tiles: pad by 8 bf16 (16 B) so row stride is
// 516 DWORDs == 4 (mod 64 banks): the 16 lanes of a half (16 different rows,
// same k) cover banks 4l..4l+3 -> all 64 banks, conflict-free ds_load_b128.
static constexpr int LDSROW = HID + 8;    // 1032 bf16 per row

__device__ __forceinline__ v16bf load_frag_lds(const bf16_t* base, int o1, int o2) {
    union { v16bf v; v8bf h[2]; } u;
    u.h[0] = *(const v8bf*)(base + o1);
    u.h[1] = *(const v8bf*)(base + o2);
    return u.v;
}

__device__ __forceinline__ void split16(const float* wv, v16bf& hi, v16bf& lo) {
#pragma unroll
    for (int i = 0; i < 16; ++i) {
        bf16_t h = (bf16_t)wv[i];
        hi[i] = h;
        lo[i] = (bf16_t)(wv[i] - (float)h);
    }
}

// Fused  logits = X @ W^T + b  over one vocab slice.
// Emits per-row partials (running max, running sumexp, target-logit-partial)
// for its slice: part[(blockIdx.y * MROWS + row)*3 + {0,1,2}].
__global__ void kto_fused_logprob_kernel(const float* __restrict__ X,
                                         const float* __restrict__ W,
                                         const float* __restrict__ Bias,
                                         const int*   __restrict__ Tgt,
                                         float*       __restrict__ part)
{
    extern __shared__ char smem_raw[];
    bf16_t* a_hi    = (bf16_t*)smem_raw;                 // [M_TILE][LDSROW]
    bf16_t* a_lo    = a_hi + M_TILE * LDSROW;            // [M_TILE][LDSROW]
    int*    tg      = (int*)(a_lo + M_TILE * LDSROW);    // [M_TILE]
    float*  tgt_lds = (float*)(tg + M_TILE);             // [M_TILE] target logit (hit-once)
    float*  red     = tgt_lds + M_TILE;                  // [M_TILE][WAVES][2]

    const int tid     = threadIdx.x;
    const int rowbase = blockIdx.x * M_TILE;

    // ---- Phase 0: split X rows into bf16 hi/lo in LDS (done once per WG) ----
    for (int idx = tid; idx < M_TILE * HID; idx += blockDim.x) {
        const int r = idx >> 10;          // HID == 1024
        const int k = idx & (HID - 1);
        float v = X[(size_t)(rowbase + r) * HID + k];
        bf16_t h = (bf16_t)v;
        a_hi[r * LDSROW + k] = h;
        a_lo[r * LDSROW + k] = (bf16_t)(v - (float)h);
    }
    if (tid < M_TILE) {
        tg[tid]      = Tgt[rowbase + tid];
        tgt_lds[tid] = 0.0f;
    }
    __syncthreads();

    const int wave = tid >> 5;
    const int lane = tid & 31;
    const int ln   = lane & 15;
    const int hi16 = lane >> 4;   // which 16-lane half
    const int sel  = hi16 * 8;    // K-chunk select per ISA 7.12.2 16-bit layout

    // Per-lane target registers: C-layout row for (tile t, acc elem e)
    int tgt_reg[2][8];
#pragma unroll
    for (int t = 0; t < 2; ++t)
#pragma unroll
        for (int e = 0; e < 8; ++e)
            tgt_reg[t][e] = tg[t * 16 + hi16 * 8 + e];

    // Online softmax state per (tile, acc element): this lane's column subset
    float mrun[2][8], srun[2][8];
#pragma unroll
    for (int t = 0; t < 2; ++t)
#pragma unroll
        for (int e = 0; e < 8; ++e) { mrun[t][e] = -3.0e38f; srun[t][e] = 0.f; }

    // A fragment row pointers (fixed for the whole kernel)
    const bf16_t* a0h = a_hi + (size_t)(ln)      * LDSROW;
    const bf16_t* a0l = a_lo + (size_t)(ln)      * LDSROW;
    const bf16_t* a1h = a_hi + (size_t)(16 + ln) * LDSROW;
    const bf16_t* a1l = a_lo + (size_t)(16 + ln) * LDSROW;

    const int ni_begin = blockIdx.y * ITERS_PER_SPLIT;
    const int ni_end   = ni_begin + ITERS_PER_SPLIT;

    for (int ni = ni_begin; ni < ni_end; ++ni) {
        const int vtile = ni * N_PER_ITER + wave * NPW;
        const int col   = vtile + ln;                      // this lane's vocab column
        const float* wrow = W + (size_t)col * HID;

        // Warm the next column tile of W (global_prefetch_b8; speculative, safe OOB)
        if (ni + 1 < ni_end)
            __builtin_prefetch(wrow + (size_t)N_PER_ITER * HID, 0, 1);

        // Fold per-column bias into the accumulator init: the load issues here
        // and its latency is hidden behind the whole K loop.
        const float bcol = Bias[col];
        v8f acc0, acc1;
#pragma unroll
        for (int i = 0; i < 8; ++i) { acc0[i] = bcol; acc1[i] = bcol; }

        for (int k0 = 0; k0 < HID; k0 += 32) {
            // --- B fragment: 16 fp32 from this lane's W row, split hi/lo ---
            alignas(16) float wv[16];
            *(f4*)(wv + 0)  = *(const f4*)(wrow + k0 + sel);
            *(f4*)(wv + 4)  = *(const f4*)(wrow + k0 + sel + 4);
            *(f4*)(wv + 8)  = *(const f4*)(wrow + k0 + 16 + sel);
            *(f4*)(wv + 12) = *(const f4*)(wrow + k0 + 16 + sel + 4);
            v16bf bhi, blo;
            split16(wv, bhi, blo);

            // --- A fragments from LDS (bank-conflict-free, see LDSROW) ---
            v16bf f0h = load_frag_lds(a0h, k0 + sel, k0 + 16 + sel);
            v16bf f0l = load_frag_lds(a0l, k0 + sel, k0 + 16 + sel);
            v16bf f1h = load_frag_lds(a1h, k0 + sel, k0 + 16 + sel);
            v16bf f1l = load_frag_lds(a1l, k0 + sel, k0 + 16 + sel);

            // --- bf16x3 split-product WMMA: hi*hi + hi*lo + lo*hi ---
            acc0 = __builtin_amdgcn_wmma_f32_16x16x32_bf16(false, f0h, false, bhi, (short)0, acc0, false, false);
            acc0 = __builtin_amdgcn_wmma_f32_16x16x32_bf16(false, f0h, false, blo, (short)0, acc0, false, false);
            acc0 = __builtin_amdgcn_wmma_f32_16x16x32_bf16(false, f0l, false, bhi, (short)0, acc0, false, false);
            acc1 = __builtin_amdgcn_wmma_f32_16x16x32_bf16(false, f1h, false, bhi, (short)0, acc1, false, false);
            acc1 = __builtin_amdgcn_wmma_f32_16x16x32_bf16(false, f1h, false, blo, (short)0, acc1, false, false);
            acc1 = __builtin_amdgcn_wmma_f32_16x16x32_bf16(false, f1l, false, bhi, (short)0, acc1, false, false);
        }

        // --- Online logsumexp epilogue for this column tile ---
        bool anyhit = false;
#pragma unroll
        for (int t = 0; t < 2; ++t) {
            v8f& acc = t ? acc1 : acc0;
#pragma unroll
            for (int e = 0; e < 8; ++e) {
                const float v  = acc[e];
                const float nm = fmaxf(mrun[t][e], v);
                srun[t][e] = srun[t][e] * __expf(mrun[t][e] - nm) + __expf(v - nm);
                mrun[t][e] = nm;
                anyhit = anyhit || (tgt_reg[t][e] == col);
            }
        }
        // Target gather: hit-once per row over the whole dispatch; the guard
        // lets the wave skip the exec-masked stores ~99% of iterations.
        if (anyhit) {
#pragma unroll
            for (int t = 0; t < 2; ++t) {
                v8f& acc = t ? acc1 : acc0;
#pragma unroll
                for (int e = 0; e < 8; ++e)
                    if (tgt_reg[t][e] == col)
                        tgt_lds[t * 16 + hi16 * 8 + e] = acc[e];
            }
        }
    }

    // ---- Lane reduction: combine the 16 lanes of each half (same row, different cols) ----
#pragma unroll
    for (int t = 0; t < 2; ++t) {
#pragma unroll
        for (int e = 0; e < 8; ++e) {
            float mv = mrun[t][e], sv = srun[t][e];
#pragma unroll
            for (int off = 1; off < 16; off <<= 1) {
                float mo = __shfl_xor(mv, off, 32);
                float so = __shfl_xor(sv, off, 32);
                float nm = fmaxf(mv, mo);
                sv = sv * __expf(mv - nm) + so * __expf(mo - nm);
                mv = nm;
            }
            if (ln == 0) {
                const int row_local = t * 16 + hi16 * 8 + e;
                float* p = red + (row_local * WAVES + wave) * 2;
                p[0] = mv; p[1] = sv;
            }
        }
    }
    __syncthreads();

    // ---- Cross-wave combine; one thread per row; emit slice partial ----
    if (tid < M_TILE) {
        float mv = -3.0e38f, sv = 0.f;
        for (int w2 = 0; w2 < WAVES; ++w2) {
            const float* p = red + (tid * WAVES + w2) * 2;
            float mo = p[0], so = p[1];
            float nm = fmaxf(mv, mo);
            sv = sv * __expf(mv - nm) + so * __expf(mo - nm);
            mv = nm;
        }
        float* p = part + ((size_t)blockIdx.y * MROWS + rowbase + tid) * 3;
        p[0] = mv;
        p[1] = sv;
        p[2] = tgt_lds[tid];   // 0 if this slice didn't contain the target column
    }
}

// Combine VSPLIT slice-partials of one row into a token log-prob.
__device__ __forceinline__ float combine_row(const float* __restrict__ part, int row) {
    float m = -3.0e38f, s = 0.f, tl = 0.f;
#pragma unroll
    for (int sp = 0; sp < VSPLIT; ++sp) {
        const float* p = part + ((size_t)sp * MROWS + row) * 3;
        const float mo = p[0], so = p[1];
        const float nm = fmaxf(m, mo);
        s = s * __expf(m - nm) + so * __expf(mo - nm);
        m = nm;
        tl += p[2];
    }
    return tl - (m + __logf(s));
}

// Reduce per-row partials into (loss, chosen_rewards_sum, rejected_rewards_sum)
__global__ void kto_finalize_kernel(const float* __restrict__ part_pol,
                                    const float* __restrict__ part_ref,
                                    const int*   __restrict__ Tgt,
                                    const int*   __restrict__ pref,
                                    const float* __restrict__ kl,
                                    float*       __restrict__ out)
{
    __shared__ float partial[256];
    const int tid = threadIdx.x;     // 256 threads: 4 batches x 64 lanes
    const int b   = tid >> 6;
    const int l   = tid & 63;
    float s = 0.f;
    for (int j = l; j < SEQ; j += 64) {
        const int r = b * SEQ + j;
        if (Tgt[r] != IGNORE_INDEX) {
            const float lp = combine_row(part_pol, r);
            const float lr = combine_row(part_ref, r);
            s += lp - lr;                 // logratio contribution
        }
    }
    partial[tid] = s;
    __syncthreads();
    for (int off = 32; off >= 1; off >>= 1) {
        if (l < off) partial[tid] += partial[tid + off];
        __syncthreads();
    }
    if (tid == 0) {
        float loss = 0.f, ch = 0.f, rj = 0.f;
        for (int bb = 0; bb < BATCH; ++bb) {
            const float lr   = partial[bb * 64];
            const float mult = pref[bb] ? 1.f : -1.f;
            const float z    = KTO_BETA * (lr - kl[bb]) * mult;
            const float sig  = 1.f / (1.f + __expf(-z));
            loss += 1.f - sig;
            const float rw = KTO_BETA * lr;
            if (pref[bb]) ch += rw; else rj += rw;
        }
        out[0] = loss / (float)BATCH;
        out[1] = ch;
        out[2] = rj;
    }
}

extern "C" void kernel_launch(void* const* d_in, const int* in_sizes, int n_in,
                              void* d_out, int out_size, void* d_ws, size_t ws_size,
                              hipStream_t stream) {
    const float* X    = (const float*)d_in[0];   // (B,T,H)
    const float* Wp   = (const float*)d_in[1];   // (V,H)
    const float* Bp   = (const float*)d_in[2];   // (V,)
    const int*   Tg   = (const int*)  d_in[3];   // (B,T)
    const int*   Pref = (const int*)  d_in[4];   // (B,)
    const float* Xr   = (const float*)d_in[5];
    const float* Wr   = (const float*)d_in[6];
    const float* Br   = (const float*)d_in[7];
    const float* Kl   = (const float*)d_in[8];

    float* ws       = (float*)d_ws;
    const size_t run_elems = (size_t)VSPLIT * MROWS * 3;   // 30720 floats per run
    float* part_pol = ws;
    float* part_ref = ws + run_elems;

    const size_t lds_bytes = (size_t)(2 * M_TILE * LDSROW) * sizeof(bf16_t)  // a_hi/a_lo (padded)
                           + (size_t)M_TILE * sizeof(int)                    // tg
                           + (size_t)M_TILE * sizeof(float)                  // tgt_lds
                           + (size_t)M_TILE * WAVES * 2 * sizeof(float);     // red

    dim3 grid(MROWS / M_TILE, VSPLIT);  // 64 x 5 = 320 workgroups
    dim3 block(WAVES * 32);             // 512 threads (16 wave32 waves)

    // Two sequential launches so each 131 MB weight matrix stays L2-resident
    kto_fused_logprob_kernel<<<grid, block, lds_bytes, stream>>>(X,  Wp, Bp, Tg, part_pol);
    kto_fused_logprob_kernel<<<grid, block, lds_bytes, stream>>>(Xr, Wr, Br, Tg, part_ref);
    kto_finalize_kernel<<<1, 256, 0, stream>>>(part_pol, part_ref, Tg, Pref, Kl, (float*)d_out);
}
